// CNN_LSTM_3358664425538
// MI455X (gfx1250) — compile-verified, matrix operand order checked by reading, not run on hardware
//
#include <hip/hip_runtime.h>
#include <hip/hip_bf16.h>

typedef __attribute__((ext_vector_type(16))) _Float16 v16h;
typedef __attribute__((ext_vector_type(8)))  _Float16 v8h;
typedef __attribute__((ext_vector_type(2)))  _Float16 v2h;
typedef __attribute__((ext_vector_type(8)))  float    v8f;

#define BT    8192   // T*B scan steps
#define HID   196    // hidden size
#define GATE  784    // 4*HID
#define KP    224    // K padded to multiple of 32 for WMMA
#define WROW  198    // padded Whh row length (halves); 99 dwords (odd) -> bank-conflict free

// ---------------------------------------------------------------------------
// utility kernels
// ---------------------------------------------------------------------------
__global__ void zero_half_kernel(_Float16* p, int n) {
    int i = blockIdx.x * blockDim.x + threadIdx.x;
    if (i < n) p[i] = (_Float16)0.f;
}

// W: [GATE][HID] f32 row-major  ->  Wh: [GATE][KP] f16, zero-padded K
__global__ void pack_w_kernel(const float* __restrict__ W, _Float16* __restrict__ Wh) {
    int i = blockIdx.x * blockDim.x + threadIdx.x;
    if (i >= GATE * KP) return;
    int r = i / KP, k = i % KP;
    Wh[i] = (k < HID) ? (_Float16)W[r * HID + k] : (_Float16)0.f;
}

// ---------------------------------------------------------------------------
// CNN front-end: conv1+bn+relu+pool, conv2+bn+relu+pool -> 196 f16 feats/sample
// one block per sample s = b*64 + t ; feats row = t*128 + b (time-major)
// ---------------------------------------------------------------------------
__global__ void conv_feats_kernel(const float* __restrict__ x,
                                  const float* __restrict__ w1, const float* __restrict__ b1,
                                  const float* __restrict__ g1, const float* __restrict__ be1,
                                  const float* __restrict__ m1, const float* __restrict__ v1,
                                  const float* __restrict__ w2, const float* __restrict__ b2,
                                  const float* __restrict__ g2, const float* __restrict__ be2,
                                  const float* __restrict__ m2, const float* __restrict__ v2,
                                  _Float16* __restrict__ featsH) {
    __shared__ float xs[784];        // 28x28 input
    __shared__ float a1[4 * 784];    // conv1 activations 4x28x28
    __shared__ float p1[4 * 196];    // pooled 4x14x14
    __shared__ float a2[4 * 196];    // conv2 activations 4x14x14
    __shared__ float w1s[36], w2s[144];
    __shared__ float s1[4], o1[4], s2[4], o2[4], b1s[4], b2s[4];

    const int tid = threadIdx.x;
    const int s = blockIdx.x;

    for (int i = tid; i < 36; i += 256) w1s[i] = w1[i];
    for (int i = tid; i < 144; i += 256) w2s[i] = w2[i];
    if (tid < 4) {
        float sc1 = g1[tid] * rsqrtf(v1[tid] + 1e-5f);
        s1[tid] = sc1; o1[tid] = be1[tid] - m1[tid] * sc1; b1s[tid] = b1[tid];
        float sc2 = g2[tid] * rsqrtf(v2[tid] + 1e-5f);
        s2[tid] = sc2; o2[tid] = be2[tid] - m2[tid] * sc2; b2s[tid] = b2[tid];
    }
    const float* xp = x + (size_t)s * 784;
    for (int i = tid; i < 784; i += 256) xs[i] = xp[i];
    __syncthreads();

    // conv1 (1->4, 3x3, pad 1) + bn + relu : 4x28x28
    for (int idx = tid; idx < 4 * 784; idx += 256) {
        int c = idx / 784, p = idx % 784, i = p / 28, j = p % 28;
        float acc = b1s[c];
        #pragma unroll
        for (int di = 0; di < 3; di++) {
            int ii = i + di - 1; if (ii < 0 || ii >= 28) continue;
            #pragma unroll
            for (int dj = 0; dj < 3; dj++) {
                int jj = j + dj - 1; if (jj < 0 || jj >= 28) continue;
                acc = fmaf(xs[ii * 28 + jj], w1s[c * 9 + di * 3 + dj], acc);
            }
        }
        float y = fmaf(acc, s1[c], o1[c]);
        a1[idx] = y > 0.f ? y : 0.f;
    }
    __syncthreads();
    // pool1 -> 4x14x14
    for (int idx = tid; idx < 4 * 196; idx += 256) {
        int c = idx / 196, p = idx % 196, i = p / 14, j = p % 14;
        const float* a = &a1[c * 784 + (2 * i) * 28 + 2 * j];
        p1[idx] = fmaxf(fmaxf(a[0], a[1]), fmaxf(a[28], a[29]));
    }
    __syncthreads();
    // conv2 (4->4, 3x3, pad 1) + bn + relu : 4x14x14
    for (int idx = tid; idx < 4 * 196; idx += 256) {
        int co = idx / 196, p = idx % 196, i = p / 14, j = p % 14;
        float acc = b2s[co];
        #pragma unroll
        for (int ci = 0; ci < 4; ci++) {
            #pragma unroll
            for (int di = 0; di < 3; di++) {
                int ii = i + di - 1; if (ii < 0 || ii >= 14) continue;
                #pragma unroll
                for (int dj = 0; dj < 3; dj++) {
                    int jj = j + dj - 1; if (jj < 0 || jj >= 14) continue;
                    acc = fmaf(p1[ci * 196 + ii * 14 + jj], w2s[(co * 4 + ci) * 9 + di * 3 + dj], acc);
                }
            }
        }
        float y = fmaf(acc, s2[co], o2[co]);
        a2[idx] = y > 0.f ? y : 0.f;
    }
    __syncthreads();
    // pool2 -> 4x7x7 = 196 feats; store f16, K-padded; time-major row
    int b = s / 64, t = s % 64;
    _Float16* fr = featsH + (size_t)(t * 128 + b) * KP;
    for (int idx = tid; idx < KP; idx += 256) {
        if (idx < 196) {
            int c = idx / 49, p = idx % 49, i = p / 7, j = p % 7;
            const float* a = &a2[c * 196 + (2 * i) * 14 + 2 * j];
            fr[idx] = (_Float16)fmaxf(fmaxf(a[0], a[1]), fmaxf(a[14], a[15]));
        } else {
            fr[idx] = (_Float16)0.f;
        }
    }
}

// ---------------------------------------------------------------------------
// xW = A[8192][KP](f16) @ Bw[GATE][KP](f16)^T + (bih+bhh), f32 out [8192][GATE]
// one wave32 per 16x16 tile, WMMA f32_16x16x32_f16, K looped 7x32
// ---------------------------------------------------------------------------
__global__ void gemm_xw_kernel(const _Float16* __restrict__ A,
                               const _Float16* __restrict__ Bw,
                               const float* __restrict__ bih,
                               const float* __restrict__ bhh,
                               float* __restrict__ xW) {
    const int lane = threadIdx.x;        // 0..31
    const int hi   = lane >> 4;          // half-wave
    const int l16  = lane & 15;
    const int m0 = blockIdx.x * 16;
    const int n0 = blockIdx.y * 16;
    const _Float16* Ar = A  + (size_t)(m0 + l16) * KP;
    const _Float16* Br = Bw + (size_t)(n0 + l16) * KP;
    v8f c = {};
    #pragma unroll
    for (int ks = 0; ks < 7; ks++) {
        // ISA 16-bit A layout: lanes 0-15 hold K 0..7 & 16..23; lanes 16-31 hold K 8..15 & 24..31
        int kb = ks * 32 + hi * 8;
        v8h a0 = *(const v8h*)(Ar + kb);
        v8h a1 = *(const v8h*)(Ar + kb + 16);
        v8h b0 = *(const v8h*)(Br + kb);
        v8h b1 = *(const v8h*)(Br + kb + 16);
        v16h av = __builtin_shufflevector(a0, a1, 0,1,2,3,4,5,6,7,8,9,10,11,12,13,14,15);
        v16h bv = __builtin_shufflevector(b0, b1, 0,1,2,3,4,5,6,7,8,9,10,11,12,13,14,15);
        c = __builtin_amdgcn_wmma_f32_16x16x32_f16(false, av, false, bv,
                                                   (short)0, c, false, false);
    }
    const int n = n0 + l16;
    const float bias = bih[n] + bhh[n];
    #pragma unroll
    for (int v = 0; v < 8; v++) {
        int row = m0 + v + hi * 8;   // D layout: VGPR v holds M=v (lanes 0-15), M=v+8 (16-31)
        xW[(size_t)row * GATE + n] = c[v] + bias;
    }
}

// ---------------------------------------------------------------------------
// serial LSTM scan: 8192 steps, single persistent workgroup.
// Whh kept resident in LDS as f16 (307KB -- only possible on CDNA5's 320KB LDS)
// ---------------------------------------------------------------------------
__global__ void lstm_scan_kernel(const float* __restrict__ xW,    // [BT][GATE]
                                 const float* __restrict__ Whh,   // [GATE][HID] f32
                                 _Float16* __restrict__ ysH,      // [BT][KP] f16 or nullptr
                                 float* __restrict__ lastRows,    // [128][HID] or nullptr
                                 float* __restrict__ hOut,        // [HID]
                                 float* __restrict__ cOut) {      // [HID]
    extern __shared__ char smem[];
    _Float16* whh = (_Float16*)smem;                       // [GATE][WROW]
    float* hS = (float*)(smem + (size_t)GATE * WROW * 2);  // [HID]
    float* cS = hS + HID;                                  // [HID]
    float* gS = cS + HID;                                  // [GATE]
    const int tid = threadIdx.x;

    for (int i = tid; i < GATE * HID; i += 256) {
        int r = i / HID, k = i % HID;
        whh[r * WROW + k] = (_Float16)Whh[i];
    }
    for (int i = tid; i < HID; i += 256) { hS[i] = 0.f; cS[i] = 0.f; }
    __syncthreads();

    const int r0 = tid, r1 = tid + 256, r2 = tid + 512, r3 = tid + 768;
    const bool has3 = (r3 < GATE);
    const v2h* w0 = (const v2h*)(whh + r0 * WROW);
    const v2h* w1 = (const v2h*)(whh + r1 * WROW);
    const v2h* w2 = (const v2h*)(whh + r2 * WROW);
    const v2h* w3 = (const v2h*)(whh + (has3 ? r3 : 0) * WROW);
    const float2* h2 = (const float2*)hS;

    for (int step = 0; step < BT; step++) {
        const float* xwr = xW + (size_t)step * GATE;
        float a0 = xwr[r0], a1v = xwr[r1], a2v = xwr[r2];
        float a3v = has3 ? xwr[r3] : 0.f;
        #pragma unroll 7
        for (int i = 0; i < HID / 2; i++) {
            float2 hv = h2[i];               // LDS broadcast read
            v2h q0 = w0[i], q1 = w1[i], q2 = w2[i], q3 = w3[i];
            a0  = fmaf((float)q0.x, hv.x, a0);  a0  = fmaf((float)q0.y, hv.y, a0);
            a1v = fmaf((float)q1.x, hv.x, a1v); a1v = fmaf((float)q1.y, hv.y, a1v);
            a2v = fmaf((float)q2.x, hv.x, a2v); a2v = fmaf((float)q2.y, hv.y, a2v);
            a3v = fmaf((float)q3.x, hv.x, a3v); a3v = fmaf((float)q3.y, hv.y, a3v);
        }
        gS[r0] = a0; gS[r1] = a1v; gS[r2] = a2v;
        if (has3) gS[r3] = a3v;
        __syncthreads();
        if (tid < HID) {
            float ig = gS[tid], fg = gS[HID + tid], gg = gS[2 * HID + tid], og = gS[3 * HID + tid];
            float si = 1.f / (1.f + expf(-ig));
            float sf = 1.f / (1.f + expf(-fg));
            float so = 1.f / (1.f + expf(-og));
            float cc = sf * cS[tid] + si * tanhf(gg);
            float hh = so * tanhf(cc);
            cS[tid] = cc; hS[tid] = hh;
            if (ysH) ysH[(size_t)step * KP + tid] = (_Float16)hh;
            if (lastRows && step >= BT - 128) lastRows[(step - (BT - 128)) * HID + tid] = hh;
            if (step == BT - 1) { hOut[tid] = hh; cOut[tid] = cc; }
        }
        __syncthreads();
    }
}

// ---------------------------------------------------------------------------
// logits = lastRows @ Wl.T + bl ; copy h/c stacks
// d_out layout: logits[128*3] | h0[196] h1[196] | c0[196] c1[196]
// ---------------------------------------------------------------------------
__global__ void finalize_kernel(const float* __restrict__ lastRows,
                                const float* __restrict__ Wl,
                                const float* __restrict__ bl,
                                const float* __restrict__ hc,  // h0,h1,c0,c1
                                float* __restrict__ out) {
    const int tid = threadIdx.x;
    if (tid < 128) {
        const float* r = lastRows + tid * HID;
        #pragma unroll
        for (int o = 0; o < 3; o++) {
            float acc = bl[o];
            for (int k = 0; k < HID; k++) acc = fmaf(r[k], Wl[o * HID + k], acc);
            out[tid * 3 + o] = acc;
        }
    }
    for (int q = tid; q < 4 * HID; q += 256) out[384 + q] = hc[q];
}

// ---------------------------------------------------------------------------
extern "C" void kernel_launch(void* const* d_in, const int* in_sizes, int n_in,
                              void* d_out, int out_size, void* d_ws, size_t ws_size,
                              hipStream_t stream) {
    const float* x    = (const float*)d_in[0];
    const float* w1   = (const float*)d_in[1];
    const float* b1   = (const float*)d_in[2];
    const float* g1   = (const float*)d_in[3];
    const float* be1  = (const float*)d_in[4];
    const float* m1   = (const float*)d_in[5];
    const float* v1   = (const float*)d_in[6];
    const float* w2   = (const float*)d_in[7];
    const float* b2   = (const float*)d_in[8];
    const float* g2   = (const float*)d_in[9];
    const float* be2  = (const float*)d_in[10];
    const float* m2   = (const float*)d_in[11];
    const float* v2   = (const float*)d_in[12];
    const float* Wih0 = (const float*)d_in[13];
    const float* Whh0 = (const float*)d_in[14];
    const float* bih0 = (const float*)d_in[15];
    const float* bhh0 = (const float*)d_in[16];
    const float* Wih1 = (const float*)d_in[17];
    const float* Whh1 = (const float*)d_in[18];
    const float* bih1 = (const float*)d_in[19];
    const float* bhh1 = (const float*)d_in[20];
    const float* Wl   = (const float*)d_in[21];
    const float* bl   = (const float*)d_in[22];

    char* ws = (char*)d_ws;
    _Float16* featsH = (_Float16*)(ws + 0);          // 8192*224*2 = 3,670,016
    _Float16* ys0H   = (_Float16*)(ws + 3670016);    // 3,670,016
    _Float16* WihH0  = (_Float16*)(ws + 7340032);    // 784*224*2 = 351,232
    _Float16* WihH1  = (_Float16*)(ws + 7691264);    // 351,232
    float*    xW     = (float*)   (ws + 8042496);    // 8192*784*4 = 25,690,112
    float*    yslast = (float*)   (ws + 33732608);   // 128*196*4 = 100,352
    float*    hc     = (float*)   (ws + 33832960);   // 4*196*4

    const size_t shmem = (size_t)GATE * WROW * 2 + (size_t)(2 * HID + GATE) * sizeof(float);
    hipFuncSetAttribute((const void*)lstm_scan_kernel,
                        hipFuncAttributeMaxDynamicSharedMemorySize, (int)shmem);

    const int nW = GATE * KP;
    pack_w_kernel<<<(nW + 255) / 256, 256, 0, stream>>>(Wih0, WihH0);
    pack_w_kernel<<<(nW + 255) / 256, 256, 0, stream>>>(Wih1, WihH1);
    const int nY = BT * KP;
    zero_half_kernel<<<(nY + 255) / 256, 256, 0, stream>>>(ys0H, nY);

    conv_feats_kernel<<<BT, 256, 0, stream>>>(x, w1, b1, g1, be1, m1, v1,
                                              w2, b2, g2, be2, m2, v2, featsH);

    dim3 gg(BT / 16, GATE / 16);  // 512 x 49 wave-tiles
    gemm_xw_kernel<<<gg, 32, 0, stream>>>(featsH, WihH0, bih0, bhh0, xW);
    lstm_scan_kernel<<<1, 256, shmem, stream>>>(xW, Whh0, ys0H, nullptr, hc, hc + 2 * HID);
    gemm_xw_kernel<<<gg, 32, 0, stream>>>(ys0H, WihH1, bih1, bhh1, xW);
    lstm_scan_kernel<<<1, 256, shmem, stream>>>(xW, Whh1, nullptr, yslast, hc + HID, hc + 3 * HID);
    finalize_kernel<<<1, 256, 0, stream>>>(yslast, Wl, bl, hc, (float*)d_out);
}
